// SparseMoETransformer_56616258896420
// MI455X (gfx1250) — matrix operand order repaired and checked by reading, not compiled, not run on hardware
//
#include <hip/hip_runtime.h>

// ---------------------------------------------------------------------------
// Types for CDNA5 WMMA (gfx1250, wave32)
// ---------------------------------------------------------------------------
typedef __attribute__((ext_vector_type(16))) __bf16 v16bf;
typedef __attribute__((ext_vector_type(8)))  float  v8f;
typedef __attribute__((ext_vector_type(4)))  int    v4i;
typedef unsigned short u16;

__device__ __forceinline__ u16 f2bf_u16(float f) {
  unsigned int u = __builtin_bit_cast(unsigned int, f);
  u += 0x7FFFu + ((u >> 16) & 1u);   // round-to-nearest-even
  return (u16)(u >> 16);
}

// ---------------------------------------------------------------------------
// Async global->LDS copy (CDNA5 ASYNCcnt path), with sync fallback.
// Prototype (from hipcc diagnostic): param0 = int4 addrspace(1)* (global src),
// param1 = int4 addrspace(3)* (LDS dst), then imm offset, imm cpol.
// ---------------------------------------------------------------------------
#if defined(__AMDGCN__) && __has_builtin(__builtin_amdgcn_global_load_async_to_lds_b128)
#define SMOE_ASYNC 1
#else
#define SMOE_ASYNC 0
#endif

__device__ __forceinline__ void smoe_cp16(const u16* __restrict__ gsrc, u16* ldst) {
#if SMOE_ASYNC
  __builtin_amdgcn_global_load_async_to_lds_b128(
      (__attribute__((address_space(1))) v4i*)gsrc,
      (__attribute__((address_space(3))) v4i*)ldst, 0, 0);
#else
  *(uint4*)ldst = *(const uint4*)gsrc;
#endif
}

#if SMOE_ASYNC
# if __has_builtin(__builtin_amdgcn_s_wait_asynccnt)
#  define SMOE_WAIT_ASYNC(imm) __builtin_amdgcn_s_wait_asynccnt(imm)
# else
#  define SMOE_WAIT_ASYNC(imm) asm volatile("s_wait_asynccnt " #imm ::: "memory")
# endif
#else
# define SMOE_WAIT_ASYNC(imm) ((void)0)
#endif

// ---------------------------------------------------------------------------
// Tiled WMMA GEMM:  C[M,N] = alpha * A[M,K] @ Bt[N,K]^T (+bias) (relu?)
// A  : bf16 [M,K] row-major (lda, batch strideA via blockIdx.z)
// Bt : bf16 [N,K] row-major (ldb, strideB)  -- B stored N-major
// C  : optional f32 out; Cb: optional bf16 out; CbT: optional transposed bf16
//      out CbT[n*ldcbt + m] (used to emit V^T directly).
// Block = 128 threads = 4 waves. Block tile 128x64, K-step 32,
// double-buffered LDS fed by async global->LDS copies (6 per thread per tile).
// Wave w computes rows [w*32, w*32+32) x 64 cols -> 2x4 wmma accumulators.
//
// LDS (padded stride LDT=40 ushorts = 80B, preserves 16B alignment):
//  As[row][*]: K-groups permuted (0,2,1,3) so a lane's 16 A elements
//              (ISA 16-bit A fragment layout) are 32B contiguous at half*16.
//  Bs[col][*]: same K-group permute as As; lane's 16 B elements contiguous at half*16.
// Out-of-range rows/cols are CLAMPED (not zero-filled): a clamped A-row/B-col
// only influences C entries that the epilogue masks out, so loads stay
// unconditional (no exec-mask branching in the staging loop).
// ---------------------------------------------------------------------------
#define GEMM_BM 128
#define GEMM_BN 64
#define GEMM_BK 32
#define GEMM_LDT 40

__global__ __launch_bounds__(128) void gemm_wmma_bf16_kernel(
    const u16* __restrict__ A,  int lda,  long strideA,
    const u16* __restrict__ Bt, int ldb,  long strideB,
    float* __restrict__ C,      int ldc,  long strideC,
    u16* __restrict__ Cb,       int ldcb, long strideCb,
    u16* __restrict__ CbT,      int ldcbt,long strideCbT,
    const float* __restrict__ bias,
    int M, int N, int K, float alpha, int relu)
{
  __shared__ __align__(16) u16 As[2 * GEMM_BM * GEMM_LDT];
  __shared__ __align__(16) u16 Bs[2 * GEMM_BN * GEMM_LDT];

  const int bz = blockIdx.z;
  A  += (long)bz * strideA;
  Bt += (long)bz * strideB;
  if (C)   C   += (long)bz * strideC;
  if (Cb)  Cb  += (long)bz * strideCb;
  if (CbT) CbT += (long)bz * strideCbT;

  const int row0 = blockIdx.y * GEMM_BM;
  const int col0 = blockIdx.x * GEMM_BN;
  const int tid  = threadIdx.x;       // 0..127
  const int wave = tid >> 5;          // 0..3
  const int lane = tid & 31;
  const int half = lane >> 4;         // 0/1
  const int l16  = lane & 15;

  v8f acc[2][4];
#pragma unroll
  for (int mt = 0; mt < 2; ++mt)
#pragma unroll
    for (int t = 0; t < 4; ++t) acc[mt][t] = v8f{};

  const int gperm[4] = {0, 2, 1, 3};   // K-group permute for A fragments

  // Stage one 128x32 A tile + 64x32 B tile into LDS buffer `buf`.
  auto stage = [&](int buf, int k0) {
    u16* Asb = As + buf * (GEMM_BM * GEMM_LDT);
    u16* Bsb = Bs + buf * (GEMM_BN * GEMM_LDT);
#pragma unroll
    for (int c = 0; c < 4; ++c) {
      const int chunk = c * 128 + tid;          // 0..511
      const int r = chunk >> 2, j = chunk & 3;  // row in tile, K-group
      int gr = row0 + r; if (gr >= M) gr = M - 1;     // clamp (see note)
      smoe_cp16(A + (long)gr * lda + (k0 + j * 8),
                &Asb[r * GEMM_LDT + gperm[j] * 8]);
    }
#pragma unroll
    for (int c = 0; c < 2; ++c) {
      const int chunk = c * 128 + tid;          // 0..255
      const int r = chunk >> 2, j = chunk & 3;  // col in tile, K-group
      int gn = col0 + r; if (gn >= N) gn = N - 1;     // clamp (see note)
      smoe_cp16(Bt + (long)gn * ldb + (k0 + j * 8),
                &Bsb[r * GEMM_LDT + gperm[j] * 8]);
    }
  };

  const int nTiles = (K + GEMM_BK - 1) / GEMM_BK;
  stage(0, 0);                                   // prologue prefetch

  for (int it = 0; it < nTiles; ++it) {
    const int cur = it & 1;
    if (it + 1 < nTiles) {
      stage(cur ^ 1, (it + 1) * GEMM_BK);        // prefetch next tile
      SMOE_WAIT_ASYNC(6);                        // tile `cur` copies done
    } else {
      SMOE_WAIT_ASYNC(0);
    }
    __syncthreads();                             // publish tile `cur`

    const u16* Asb = As + cur * (GEMM_BM * GEMM_LDT);
    const u16* Bsb = Bs + cur * (GEMM_BN * GEMM_LDT);

    v16bf bfr[4];
#pragma unroll
    for (int t = 0; t < 4; ++t)
      bfr[t] = *(const v16bf*)(&Bsb[(t * 16 + l16) * GEMM_LDT + half * 16]);

#pragma unroll
    for (int mt = 0; mt < 2; ++mt) {
      const int arow = (wave << 5) + (mt << 4) + l16;
      v16bf af = *(const v16bf*)(&Asb[arow * GEMM_LDT + half * 16]);
#pragma unroll
      for (int t = 0; t < 4; ++t)
        acc[mt][t] = __builtin_amdgcn_wmma_f32_16x16x32_bf16(
            false, af, false, bfr[t], (short)0, acc[mt][t], false, false);
    }
    __syncthreads();                             // all reads of `cur` done
  }

  // ---- Epilogue: VGPR r holds M = r + 8*half, N = l16 within 16x16 tile ----
#pragma unroll
  for (int mt = 0; mt < 2; ++mt) {
    const int mbase = row0 + (wave << 5) + (mt << 4) + (half << 3);
#pragma unroll
    for (int t = 0; t < 4; ++t) {
      const int gc = col0 + (t << 4) + l16;
      if (gc >= N) continue;
      const float bval = bias ? bias[gc] : 0.0f;
#pragma unroll
      for (int r = 0; r < 8; ++r) {
        const int gr = mbase + r;
        if (gr >= M) continue;
        float v = acc[mt][t][r] * alpha + bval;
        if (relu) v = v > 0.0f ? v : 0.0f;
        if (C)   C[(long)gr * ldc + gc] = v;
        if (Cb)  Cb[(long)gr * ldcb + gc] = f2bf_u16(v);
        if (CbT) CbT[(long)gc * ldcbt + gr] = f2bf_u16(v);
      }
    }
  }
}

// ---------------------------------------------------------------------------
// Elementwise / small kernels
// ---------------------------------------------------------------------------
__global__ void smoe_f32_to_bf16(const float* __restrict__ src,
                                 u16* __restrict__ dst, long n) {
  long i = (long)blockIdx.x * blockDim.x + threadIdx.x;
  if (i < n) dst[i] = f2bf_u16(src[i]);
}

// Batched transpose + convert: dst[b][n][k] = bf16(src[b][k][n]); src [batch,K,N]
__global__ void smoe_conv_transpose(const float* __restrict__ src,
                                    u16* __restrict__ dst,
                                    int K, int N, long total) {
  long i = (long)blockIdx.x * blockDim.x + threadIdx.x;
  if (i >= total) return;
  const long per = (long)K * N;
  const long b = i / per;
  const long r = i - b * per;
  const int n = (int)(r / K);
  const int k = (int)(r - (long)n * K);
  dst[b * per + (long)n * K + k] = f2bf_u16(src[b * per + (long)k * N + n]);
}

__global__ void smoe_zero_f32(float* __restrict__ p, long n) {
  long i = (long)blockIdx.x * blockDim.x + threadIdx.x;
  if (i < n) p[i] = 0.0f;
}

__global__ void smoe_embed(const int* __restrict__ inputs,
                           const float* __restrict__ tok,
                           const float* __restrict__ pos,
                           float* __restrict__ x,
                           u16* __restrict__ xb,
                           long BT, int E, int T) {
  long i = (long)blockIdx.x * blockDim.x + threadIdx.x;
  if (i >= BT * (long)E) return;
  long row = i / E;
  int  e   = (int)(i - row * E);
  int  t   = (int)(row % T);
  int  tk  = inputs[row];
  float v = tok[(long)tk * E + e] + pos[(long)t * E + e];
  x[i]  = v;
  xb[i] = f2bf_u16(v);
}

// Causal softmax over scores [H, T, T] for one batch element, in place,
// also producing a bf16 copy (zeros in the masked region).
__global__ __launch_bounds__(256) void smoe_softmax_causal(
    float* __restrict__ S, u16* __restrict__ Sb, int T) {
  const int t = blockIdx.x, h = blockIdx.y, tid = threadIdx.x;
  float* row = S + ((long)h * T + t) * (long)T;
  u16* brow = Sb + ((long)h * T + t) * (long)T;
  const int n = t + 1;
  __shared__ float red[256];

  float mx = -3.4e38f;
  for (int s = tid; s < n; s += 256) mx = fmaxf(mx, row[s]);
  red[tid] = mx; __syncthreads();
  for (int s = 128; s > 0; s >>= 1) {
    if (tid < s) red[tid] = fmaxf(red[tid], red[tid + s]);
    __syncthreads();
  }
  mx = red[0]; __syncthreads();

  float sum = 0.0f;
  for (int s = tid; s < n; s += 256) {
    float e = __expf(row[s] - mx);
    row[s] = e;
    sum += e;
  }
  red[tid] = sum; __syncthreads();
  for (int s = 128; s > 0; s >>= 1) {
    if (tid < s) red[tid] += red[tid + s];
    __syncthreads();
  }
  const float inv = 1.0f / red[0];

  for (int s = tid; s < n; s += 256) {
    float p = row[s] * inv;
    row[s]  = p;
    brow[s] = f2bf_u16(p);
  }
  for (int s = n + tid; s < T; s += 256) {
    row[s]  = 0.0f;
    brow[s] = 0;
  }
}

// y = LayerNorm(p + q) * gamma + beta ; writes f32 y and bf16 yb. E == 512.
__global__ __launch_bounds__(256) void smoe_add_ln(
    const float* __restrict__ p, const float* __restrict__ q,
    const float* __restrict__ gamma, const float* __restrict__ beta,
    float* __restrict__ y, u16* __restrict__ yb, int E) {
  const long row = blockIdx.x;
  const int tid = threadIdx.x;
  const float* pr = p + row * (long)E;
  const float* qr = q + row * (long)E;
  __shared__ float red[256];

  float v0 = pr[tid]       + qr[tid];
  float v1 = pr[tid + 256] + qr[tid + 256];

  red[tid] = v0 + v1; __syncthreads();
  for (int s = 128; s > 0; s >>= 1) {
    if (tid < s) red[tid] += red[tid + s];
    __syncthreads();
  }
  const float mean = red[0] / (float)E; __syncthreads();

  float d0 = v0 - mean, d1 = v1 - mean;
  red[tid] = d0 * d0 + d1 * d1; __syncthreads();
  for (int s = 128; s > 0; s >>= 1) {
    if (tid < s) red[tid] += red[tid + s];
    __syncthreads();
  }
  const float rstd = rsqrtf(red[0] / (float)E + 1e-5f);

  float o0 = d0 * rstd * gamma[tid]       + beta[tid];
  float o1 = d1 * rstd * gamma[tid + 256] + beta[tid + 256];
  y[row * (long)E + tid]        = o0;
  y[row * (long)E + tid + 256]  = o1;
  yb[row * (long)E + tid]       = f2bf_u16(o0);
  yb[row * (long)E + tid + 256] = f2bf_u16(o1);
}

// probs = softmax(x1 @ Wr + br) ; NE == 8, one 64-thread block per row.
__global__ __launch_bounds__(64) void smoe_router(
    const float* __restrict__ x1, const float* __restrict__ Wr,
    const float* __restrict__ br, float* __restrict__ probs, int E, int NE) {
  const long row = blockIdx.x;
  const int tid = threadIdx.x;
  float acc[8] = {0, 0, 0, 0, 0, 0, 0, 0};
  for (int e = tid; e < E; e += 64) {
    float xv = x1[row * (long)E + e];
#pragma unroll
    for (int n = 0; n < 8; ++n) acc[n] += xv * Wr[(long)e * NE + n];
  }
  __shared__ float red[8][64];
#pragma unroll
  for (int n = 0; n < 8; ++n) red[n][tid] = acc[n];
  __syncthreads();
  __shared__ float logit[8];
  if (tid < 8) {
    float s = 0.0f;
    for (int j = 0; j < 64; ++j) s += red[tid][j];
    logit[tid] = s + br[tid];
  }
  __syncthreads();
  if (tid == 0) {
    float mx = logit[0];
    for (int n = 1; n < 8; ++n) mx = fmaxf(mx, logit[n]);
    float sum = 0.0f, ex[8];
    for (int n = 0; n < 8; ++n) { ex[n] = __expf(logit[n] - mx); sum += ex[n]; }
    float inv = 1.0f / sum;
    for (int n = 0; n < 8; ++n) probs[row * (long)NE + n] = ex[n] * inv;
  }
}

// moe += probs[:, n] * eo   (elementwise over [BT, E])
__global__ void smoe_moe_acc(float* __restrict__ moe,
                             const float* __restrict__ eo,
                             const float* __restrict__ probs,
                             int n, int E, int NE, long total) {
  long i = (long)blockIdx.x * blockDim.x + threadIdx.x;
  if (i >= total) return;
  long row = i / E;
  moe[i] += probs[row * (long)NE + n] * eo[i];
}

// In-place log_softmax over V per row.
__global__ __launch_bounds__(256) void smoe_log_softmax(
    float* __restrict__ logits, int V) {
  const long row = blockIdx.x;
  const int tid = threadIdx.x;
  float* r = logits + row * (long)V;
  __shared__ float red[256];

  float mx = -3.4e38f;
  for (int i = tid; i < V; i += 256) mx = fmaxf(mx, r[i]);
  red[tid] = mx; __syncthreads();
  for (int s = 128; s > 0; s >>= 1) {
    if (tid < s) red[tid] = fmaxf(red[tid], red[tid + s]);
    __syncthreads();
  }
  mx = red[0]; __syncthreads();

  float sum = 0.0f;
  for (int i = tid; i < V; i += 256) sum += __expf(r[i] - mx);
  red[tid] = sum; __syncthreads();
  for (int s = 128; s > 0; s >>= 1) {
    if (tid < s) red[tid] += red[tid + s];
    __syncthreads();
  }
  const float lse = mx + __logf(red[0]);

  for (int i = tid; i < V; i += 256) r[i] = r[i] - lse;
}

// ---------------------------------------------------------------------------
// Host orchestration
// ---------------------------------------------------------------------------
extern "C" void kernel_launch(void* const* d_in, const int* in_sizes, int n_in,
                              void* d_out, int out_size, void* d_ws, size_t ws_size,
                              hipStream_t stream) {
  (void)in_sizes; (void)n_in; (void)out_size; (void)ws_size;
  const int Bn = 8, T = 1024, E = 512, H = 8, Ln = 4, NE = 8, V = 10000;
  const int HD = E / H, FF = 4 * E;
  const long BT = (long)Bn * T;

  const int*   inputs = (const int*)  d_in[0];
  const float* tok    = (const float*)d_in[1];
  const float* pos    = (const float*)d_in[2];
  const float* Wq  = (const float*)d_in[3];  const float* bq  = (const float*)d_in[4];
  const float* Wk  = (const float*)d_in[5];  const float* bk  = (const float*)d_in[6];
  const float* Wv  = (const float*)d_in[7];  const float* bv  = (const float*)d_in[8];
  const float* Wo  = (const float*)d_in[9];  const float* bo  = (const float*)d_in[10];
  const float* g1  = (const float*)d_in[11]; const float* be1 = (const float*)d_in[12];
  const float* Wr  = (const float*)d_in[13]; const float* br  = (const float*)d_in[14];
  const float* W1  = (const float*)d_in[15]; const float* b1  = (const float*)d_in[16];
  const float* W2  = (const float*)d_in[17]; const float* b2  = (const float*)d_in[18];
  const float* g2  = (const float*)d_in[19]; const float* be2 = (const float*)d_in[20];
  const float* Wfc = (const float*)d_in[23]; const float* bfc = (const float*)d_in[24];

  // Bump allocator over d_ws
  size_t off = 0;
  auto alloc = [&](size_t bytes) -> void* {
    off = (off + 255) & ~(size_t)255;
    void* p = (char*)d_ws + off;
    off += bytes;
    return p;
  };

  // Weights, transposed to [N,K] bf16
  u16* wq_t  = (u16*)alloc((size_t)Ln * E * E * 2);
  u16* wk_t  = (u16*)alloc((size_t)Ln * E * E * 2);
  u16* wv_t  = (u16*)alloc((size_t)Ln * E * E * 2);
  u16* wo_t  = (u16*)alloc((size_t)Ln * E * E * 2);
  u16* w1_t  = (u16*)alloc((size_t)Ln * NE * E * FF * 2);
  u16* w2_t  = (u16*)alloc((size_t)Ln * NE * FF * E * 2);
  u16* wfc_t = (u16*)alloc((size_t)E * V * 2);

  float* xa    = (float*)alloc((size_t)BT * E * 4);
  float* xbuf  = (float*)alloc((size_t)BT * E * 4);
  u16*   x_bf  = (u16*)  alloc((size_t)BT * E * 2);
  float* x1_f  = (float*)alloc((size_t)BT * E * 4);
  u16*   x1_bf = (u16*)  alloc((size_t)BT * E * 2);
  u16*   q_bf  = (u16*)  alloc((size_t)BT * E * 2);
  u16*   k_bf  = (u16*)  alloc((size_t)BT * E * 2);
  u16*   vT_bf = (u16*)  alloc((size_t)BT * E * 2);   // [B][E][T] (V transposed)
  float* scores= (float*)alloc((size_t)H * T * T * 4);  // one batch element
  u16*   att_bf= (u16*)  alloc((size_t)H * T * T * 2);
  float* o_f   = (float*)alloc((size_t)BT * E * 4);
  u16*   o_bf  = (u16*)  alloc((size_t)BT * E * 2);
  float* a_f   = (float*)alloc((size_t)BT * E * 4);
  float* probs = (float*)alloc((size_t)BT * NE * 4);
  u16*   h_bf  = (u16*)  alloc((size_t)BT * FF * 2);
  float* eo_f  = (float*)alloc((size_t)BT * E * 4);
  float* moe_f = (float*)alloc((size_t)BT * E * 4);

  auto convT = [&](const float* src, u16* dst, int K, int N, long batch) {
    long n = batch * (long)K * N;
    smoe_conv_transpose<<<dim3((unsigned)((n + 255) / 256)), dim3(256), 0, stream>>>(
        src, dst, K, N, n);
  };
  auto conv = [&](const float* src, u16* dst, long n) {
    smoe_f32_to_bf16<<<dim3((unsigned)((n + 255) / 256)), dim3(256), 0, stream>>>(src, dst, n);
  };
  auto gemm = [&](const u16* A, int lda, long sA,
                  const u16* Bt, int ldb, long sB,
                  float* C, int ldc, long sC,
                  u16* Cb, int ldcb, long sCb,
                  u16* CbT, int ldcbt, long sCbT,
                  const float* bias, int M, int N, int K,
                  float alpha, int relu, int batch) {
    dim3 grid((N + GEMM_BN - 1) / GEMM_BN, (M + GEMM_BM - 1) / GEMM_BM, batch);
    gemm_wmma_bf16_kernel<<<grid, dim3(128), 0, stream>>>(
        A, lda, sA, Bt, ldb, sB, C, ldc, sC, Cb, ldcb, sCb, CbT, ldcbt, sCbT,
        bias, M, N, K, alpha, relu);
  };

  // Convert + transpose all weights to bf16 [N,K] (once per launch).
  convT(Wq,  wq_t,  E, E,  Ln);
  convT(Wk,  wk_t,  E, E,  Ln);
  convT(Wv,  wv_t,  E, E,  Ln);
  convT(Wo,  wo_t,  E, E,  Ln);
  convT(W1,  w1_t,  E, FF, (long)Ln * NE);
  convT(W2,  w2_t,  FF, E, (long)Ln * NE);
  convT(Wfc, wfc_t, E, V,  1);

  // Embedding: x = tok_emb[inputs] + pos_emb
  {
    long n = BT * (long)E;
    smoe_embed<<<dim3((unsigned)((n + 255) / 256)), dim3(256), 0, stream>>>(
        inputs, tok, pos, xa, x_bf, BT, E, T);
  }

  for (int l = 0; l < Ln; ++l) {
    const size_t wEE = (size_t)l * E * E;
    // Q, K projections -> bf16 [BT, E]
    gemm(x_bf, E, 0, wq_t + wEE, E, 0, nullptr, 0, 0, q_bf, E, 0,
         nullptr, 0, 0, bq + (size_t)l * E, (int)BT, E, E, 1.0f, 0, 1);
    gemm(x_bf, E, 0, wk_t + wEE, E, 0, nullptr, 0, 0, k_bf, E, 0,
         nullptr, 0, 0, bk + (size_t)l * E, (int)BT, E, E, 1.0f, 0, 1);
    // V projection, emitted TRANSPOSED per batch element: vT[b][e][t]
    gemm(x_bf, E, (long)T * E, wv_t + wEE, E, 0, nullptr, 0, 0, nullptr, 0, 0,
         vT_bf, T, (long)E * T, bv + (size_t)l * E, T, E, E, 1.0f, 0, Bn);

    // Attention, batched over heads, looped over batch element b
    for (int b = 0; b < Bn; ++b) {
      const u16* qb = q_bf + (size_t)b * T * E;
      const u16* kb = k_bf + (size_t)b * T * E;
      const u16* vb = vT_bf + (size_t)b * E * T;
      // scores[h,t,s] = (1/8) * sum_d q[t, h*64+d] * k[s, h*64+d]
      gemm(qb, E, HD, kb, E, HD,
           scores, T, (long)T * T, nullptr, 0, 0, nullptr, 0, 0, nullptr,
           T, T, HD, 0.125f, 0, H);
      smoe_softmax_causal<<<dim3(T, H), dim3(256), 0, stream>>>(scores, att_bf, T);
      // o[t, h*64+d] = sum_s att[h,t,s] * vT[h*64+d, s]
      gemm(att_bf, T, (long)T * T, vb, T, (long)HD * T,
           o_f + (size_t)b * T * E, E, HD, nullptr, 0, 0, nullptr, 0, 0, nullptr,
           T, HD, T, 1.0f, 0, H);
    }
    conv(o_f, o_bf, BT * (long)E);

    // Output projection: a = o @ Wo + bo
    gemm(o_bf, E, 0, wo_t + wEE, E, 0, a_f, E, 0, nullptr, 0, 0,
         nullptr, 0, 0, bo + (size_t)l * E, (int)BT, E, E, 1.0f, 0, 1);

    // x1 = LN(a + x)
    smoe_add_ln<<<dim3((unsigned)BT), dim3(256), 0, stream>>>(
        a_f, xa, g1 + (size_t)l * E, be1 + (size_t)l * E, x1_f, x1_bf, E);

    // Router probs
    smoe_router<<<dim3((unsigned)BT), dim3(64), 0, stream>>>(
        x1_f, Wr + (size_t)l * E * NE, br + (size_t)l * NE, probs, E, NE);

    // Dense MoE
    {
      long n = BT * (long)E;
      smoe_zero_f32<<<dim3((unsigned)((n + 255) / 256)), dim3(256), 0, stream>>>(moe_f, n);
    }
    for (int n = 0; n < NE; ++n) {
      const size_t we = ((size_t)l * NE + n);
      // h = relu(x1 @ W1[l,n] + b1[l,n]) -> bf16
      gemm(x1_bf, E, 0, w1_t + we * E * FF, E, 0,
           nullptr, 0, 0, h_bf, FF, 0, nullptr, 0, 0,
           b1 + we * FF, (int)BT, FF, E, 1.0f, /*relu=*/1, 1);
      // eo = h @ W2[l,n] + b2[l,n]
      gemm(h_bf, FF, 0, w2_t + we * FF * E, FF, 0,
           eo_f, E, 0, nullptr, 0, 0, nullptr, 0, 0,
           b2 + we * E, (int)BT, E, FF, 1.0f, 0, 1);
      long tot = BT * (long)E;
      smoe_moe_acc<<<dim3((unsigned)((tot + 255) / 256)), dim3(256), 0, stream>>>(
          moe_f, eo_f, probs, n, E, NE, tot);
    }

    // x = LN(moe + x)  (residual is the block input)
    smoe_add_ln<<<dim3((unsigned)BT), dim3(256), 0, stream>>>(
        moe_f, xa, g2 + (size_t)l * E, be2 + (size_t)l * E, xbuf, x_bf, E);

    float* tmp = xa; xa = xbuf; xbuf = tmp;   // ping-pong; x_bf tracks xa
  }

  // logits = x @ Wfc + bfc, then log_softmax in place -> d_out (f32)
  gemm(x_bf, E, 0, wfc_t, E, 0, (float*)d_out, V, 0, nullptr, 0, 0,
       nullptr, 0, 0, bfc, (int)BT, V, E, 1.0f, 0, 1);
  smoe_log_softmax<<<dim3((unsigned)BT), dim3(256), 0, stream>>>((float*)d_out, V);
}
